// STBlock_386547057198
// MI455X (gfx1250) — matrix operand order repaired and checked by reading, not compiled, not run on hardware
//
#include <hip/hip_runtime.h>
#include <math.h>

// ---------------------------------------------------------------------------
// STBlock for MI455X (gfx1250, wave32). All GEMMs and both attention einsums
// run on v_wmma_f32_16x16x32_f16 (f16 in, f32 accumulate). fp32 master
// activations for residuals / LayerNorm. GEMM weight panels are staged into
// LDS by the Tensor Data Mover (TENSOR_LOAD_TO_LDS + s_wait_tensorcnt).
// B=8 N=256 T=64 C=64 H=4 dh=16 FF=256. M = B*N*T = 131072 rows.
// ---------------------------------------------------------------------------

typedef __attribute__((ext_vector_type(16))) _Float16 v16h;
typedef __attribute__((ext_vector_type(8)))  float    v8f;
typedef __attribute__((ext_vector_type(4)))  unsigned uv4;
typedef __attribute__((ext_vector_type(8)))  int      iv8;
typedef __attribute__((ext_vector_type(4)))  int      iv4;

#if defined(__has_builtin)
#if __has_builtin(__builtin_amdgcn_tensor_load_to_lds)
#define HAVE_TDM 1
#endif
#endif
#if defined(__has_include)
#if __has_include(<hip/amd_detail/amd_gfx1250_TDM.h>)
#define TDM_6ARG 1
#endif
#endif

union FragH { v16h v; unsigned u[8]; _Float16 h[16]; };

__device__ inline v8f v8f_zero() {
  v8f z = {0.f,0.f,0.f,0.f,0.f,0.f,0.f,0.f};
  return z;
}

#define WMMA16(A,B,C) __builtin_amdgcn_wmma_f32_16x16x32_f16(false,(A),false,(B),(short)0,(C),false,false)

// A-fragment (16x32 f16): lane L -> row = row16 + (L&15); K halves per ISA:
//   kb = (L<16)?0:8 ; VGPR j<4 holds K = kb+2j,kb+2j+1 ; VGPR 4+j holds K=16+kb+2j,...
__device__ inline void load_afrag(FragH& f, const _Float16* base, int rowStride,
                                  int row16, int lane) {
  const int r  = row16 + (lane & 15);
  const int kb = (lane < 16) ? 0 : 8;
  const _Float16* p = base + r * rowStride;
#pragma unroll
  for (int j = 0; j < 4; ++j) {
    f.u[j]     = *(const unsigned*)(p + kb + 2 * j);
    f.u[4 + j] = *(const unsigned*)(p + 16 + kb + 2 * j);
  }
}

// B-fragment (32x16 f16, K-major storage): lane L -> N = colBase + (L&15);
// kb = (L<16)?0:8 ; element s<8 holds K = kb+s, element 8+s holds K = 16+kb+s.
__device__ inline void load_bfrag(FragH& f, const _Float16* base, int rowStride,
                                  int colBase, int lane) {
  const int kb = (lane < 16) ? 0 : 8;
  const _Float16* p = base + colBase + (lane & 15);
#pragma unroll
  for (int s = 0; s < 8; ++s) {
    f.h[s]     = p[(kb + s) * rowStride];
    f.h[8 + s] = p[(16 + kb + s) * rowStride];
  }
}

// ---------------------------------------------------------------------------
// TDM: DMA a 2D f16 weight tile [K rows x 64 cols] (row stride Ncols) into
// LDS at byte offset ldsOff. Issued by a single wave; completion via
// TENSORcnt. D# layout per CDNA5 ISA ch.8 (group0 + group1, groups 2/3 zero).
// ---------------------------------------------------------------------------
#ifdef HAVE_TDM
__device__ inline void tdm_load_panel(const _Float16* Bw, unsigned ldsOff,
                                      int K, int Ncols, int colBase) {
  unsigned long long ga = (unsigned long long)(const void*)(Bw + colBase);
  uv4 g0;
  g0[0] = 1u;                                        // count=1, user desc
  g0[1] = ldsOff;                                    // lds_addr
  g0[2] = (unsigned)(ga & 0xffffffffu);              // global_addr[31:0]
  g0[3] = (unsigned)((ga >> 32) & 0x01ffffffu) | 0x80000000u;  // addr[56:32]|type=2
  iv8 g1;
  g1[0] = 0x00010000;                                // data_size=1 (2 bytes)
  g1[1] = (int)((unsigned)Ncols << 16);              // tensor_dim0[15:0]
  g1[2] = (int)((((unsigned)Ncols) >> 16) | ((unsigned)K << 16));  // dim0 hi|dim1 lo
  g1[3] = (int)((((unsigned)K) >> 16) | (64u << 16));              // dim1 hi|tile_dim0=64
  g1[4] = (int)(unsigned)K;                          // tile_dim1=K, tile_dim2=0
  g1[5] = (int)(unsigned)Ncols;                      // tensor_dim0_stride[31:0]
  g1[6] = 0;
  g1[7] = 0;
  iv4 gz = {0, 0, 0, 0};
#ifdef TDM_6ARG
  iv8 gz8 = {0, 0, 0, 0, 0, 0, 0, 0};
  __builtin_amdgcn_tensor_load_to_lds(g0, g1, gz, gz, gz8, 0);
#else
  __builtin_amdgcn_tensor_load_to_lds(g0, g1, gz, gz, 0);
#endif
  __builtin_amdgcn_s_wait_tensorcnt(0);
}
#endif

// ---------------------------------------------------------------------------
// Generic WMMA GEMM: out[M,N] = A[M,K](f16) x B[K,N](f16) (+bias)(relu?)
// Block tile 128x64. Weight panel B[K x 64] DMA'd once into LDS by the TDM;
// A staged in 32-wide K chunks with vector loads + prefetch of next chunk.
// 8 waves/block, wave w owns rows [16w,16w+16), 4 N-tiles each.
// grid = (M/128, N/64). K <= 256.
// ---------------------------------------------------------------------------
__global__ __launch_bounds__(256) void gemm_f16_wmma(
    const _Float16* __restrict__ A, const _Float16* __restrict__ Bw,
    const float* __restrict__ bias, float* __restrict__ outF,
    _Float16* __restrict__ outH, int Mrows, int K, int Ncols, int relu) {
  __shared__ char smem[8192 + 32768];           // As 128x32 f16 | Bs Kx64 f16
  _Float16* As = (_Float16*)smem;
  _Float16* Bs = (_Float16*)(smem + 8192);
  const int tid = threadIdx.x;
  const int wave = tid >> 5, lane = tid & 31;
  const int rowBase = blockIdx.x * 128;
  const int colBase = blockIdx.y * 64;

#ifdef HAVE_TDM
  if (tid < 32) {                                // single wave issues the DMA
    tdm_load_panel(Bw, 8192u, K, Ncols, colBase);
  }
#else
  for (int u = tid; u < (K * 64) / 2; u += 256) {
    int kk = u >> 5, cp = u & 31;
    ((unsigned*)Bs)[u] =
        *(const unsigned*)(Bw + (size_t)kk * Ncols + colBase + cp * 2);
  }
#endif

  v8f acc[4];
#pragma unroll
  for (int i = 0; i < 4; ++i) acc[i] = v8f_zero();

  bool firstBar = true;
  for (int k0 = 0; k0 < K; k0 += 32) {
    // Stage A 128x32 (2048 dwords), 8 per thread.
#pragma unroll
    for (int i = 0; i < 8; ++i) {
      int u = tid + i * 256;
      int r = u >> 4, cp = u & 15;
      ((unsigned*)As)[u] =
          *(const unsigned*)(A + (size_t)(rowBase + r) * K + k0 + cp * 2);
    }
    if (k0 + 32 < K) {  // prefetch next A chunk (global_prefetch_b8)
      __builtin_prefetch(A + (size_t)(rowBase + (tid >> 1)) * K + k0 + 32, 0, 0);
    }
    __syncthreads();    // first one also covers TDM panel visibility
    (void)firstBar;

    FragH af;
    load_afrag(af, As, 32, wave * 16, lane);
#pragma unroll
    for (int nt = 0; nt < 4; ++nt) {
      FragH bf;
      load_bfrag(bf, Bs + (size_t)k0 * 64, 64, nt * 16, lane);
      acc[nt] = WMMA16(af.v, bf.v, acc[nt]);
    }
    __syncthreads();
  }

  // Epilogue. C layout: VGPR j -> row j (lanes 0-15) / j+8 (lanes 16-31),
  // col = lane&15 within the tile.
  const int rofs = (lane < 16) ? 0 : 8;
  float res[4][8];
#pragma unroll
  for (int nt = 0; nt < 4; ++nt) {
    const float bv = bias ? bias[colBase + nt * 16 + (lane & 15)] : 0.f;
#pragma unroll
    for (int j = 0; j < 8; ++j) {
      float v = acc[nt][j] + bv;
      if (relu) v = fmaxf(v, 0.f);
      res[nt][j] = v;
    }
  }
  if (outF) {
#pragma unroll
    for (int nt = 0; nt < 4; ++nt)
#pragma unroll
      for (int j = 0; j < 8; ++j) {
        const int row = rowBase + wave * 16 + j + rofs;
        outF[(size_t)row * Ncols + colBase + nt * 16 + (lane & 15)] = res[nt][j];
      }
  }
  if (outH) {
#pragma unroll
    for (int nt = 0; nt < 4; ++nt)
#pragma unroll
      for (int j = 0; j < 8; ++j) {
        const int row = rowBase + wave * 16 + j + rofs;
        outH[(size_t)row * Ncols + colBase + nt * 16 + (lane & 15)] =
            (_Float16)res[nt][j];
      }
  }
}

// ---------------------------------------------------------------------------
// Fused spatial attention, one block per (b,h,t). N=256 queries/keys, dh=16
// (padded to 32 so one 16x16x32 WMMA makes a 16x16 score tile). 8 waves,
// wave w owns query rows [32w,32w+32) as two 16-row M-tiles. Online softmax;
// P re-laid-out C->A through a per-wave LDS pad; PxV via WMMA.
// ---------------------------------------------------------------------------
__global__ __launch_bounds__(256) void attn_spatial_wmma(
    const _Float16* __restrict__ q, const _Float16* __restrict__ k,
    const _Float16* __restrict__ v, const float* __restrict__ biasH,
    _Float16* __restrict__ ctx) {
  __shared__ _Float16 Qs[256 * 32];   // [n][dh padded 32]
  __shared__ _Float16 KsT[32 * 256];  // [dh padded 32][n]
  __shared__ _Float16 Vs[256 * 16];   // [n][dh]
  __shared__ _Float16 Ps[8][32 * 32]; // per-wave prob pad [row][key]

  const int bidx = blockIdx.x;
  const int t = bidx & 63, h = (bidx >> 6) & 3, b = bidx >> 8;
  const int tid = threadIdx.x, wave = tid >> 5, lane = tid & 31;

  {  // each thread loads node row n = tid
    const int n = tid;
    const size_t base = (((size_t)(b * 256 + n)) * 64 + t) * 64 + h * 16;
#pragma unroll
    for (int d2 = 0; d2 < 8; ++d2) {
      ((unsigned*)&Qs[n * 32])[d2]     = *(const unsigned*)(q + base + 2 * d2);
      ((unsigned*)&Qs[n * 32])[8 + d2] = 0u;
      ((unsigned*)&Vs[n * 16])[d2]     = *(const unsigned*)(v + base + 2 * d2);
    }
#pragma unroll
    for (int d = 0; d < 16; ++d) {
      KsT[d * 256 + n]        = k[base + d];
      KsT[(16 + d) * 256 + n] = (_Float16)0.f;
    }
  }
  __syncthreads();

  FragH qa[2];
  load_afrag(qa[0], Qs, 32, wave * 32, lane);
  load_afrag(qa[1], Qs, 32, wave * 32 + 16, lane);

  v8f O[2] = {v8f_zero(), v8f_zero()};
  float mrun[2][8], srun[2][8];
#pragma unroll
  for (int mt = 0; mt < 2; ++mt)
#pragma unroll
    for (int j = 0; j < 8; ++j) { mrun[mt][j] = -1e30f; srun[mt][j] = 0.f; }

  const float scale = 0.25f;  // 1/sqrt(dh=16)
  const int rofs = (lane < 16) ? 0 : 8;
  const int colLane = lane & 15;

  for (int c = 0; c < 8; ++c) {  // 32-key chunks
    FragH kb0, kb1, vb;
    load_bfrag(kb0, KsT, 256, c * 32, lane);
    load_bfrag(kb1, KsT, 256, c * 32 + 16, lane);
    load_bfrag(vb, Vs + (size_t)c * 32 * 16, 16, 0, lane);

#pragma unroll
    for (int mt = 0; mt < 2; ++mt) {
      v8f S0 = WMMA16(qa[mt].v, kb0.v, v8f_zero());
      v8f S1 = WMMA16(qa[mt].v, kb1.v, v8f_zero());
#pragma unroll
      for (int j = 0; j < 8; ++j) {
        const int nrow = wave * 32 + mt * 16 + j + rofs;
        const size_t bb = ((size_t)(h * 256 + nrow)) * 256 + c * 32 + colLane;
        S0[j] = S0[j] * scale + biasH[bb];
        S1[j] = S1[j] * scale + biasH[bb + 16];
      }
#pragma unroll
      for (int j = 0; j < 8; ++j) {
        float mx = fmaxf(S0[j], S1[j]);
        for (int off = 8; off; off >>= 1) mx = fmaxf(mx, __shfl_xor(mx, off, 16));
        const float f0 = fmaxf(mrun[mt][j], mx);
        const float f = expf(mrun[mt][j] - f0);
        mrun[mt][j] = f0;
        S0[j] = expf(S0[j] - f0);
        S1[j] = expf(S1[j] - f0);
        float rs = S0[j] + S1[j];
        for (int off = 8; off; off >>= 1) rs += __shfl_xor(rs, off, 16);
        srun[mt][j] = srun[mt][j] * f + rs;
        O[mt][j] *= f;
        const int rl = mt * 16 + j + rofs;
        Ps[wave][rl * 32 + colLane]      = (_Float16)S0[j];
        Ps[wave][rl * 32 + 16 + colLane] = (_Float16)S1[j];
      }
    }
    // per-wave LDS region: DS ops in-order within a wave, no barrier needed
#pragma unroll
    for (int mt = 0; mt < 2; ++mt) {
      FragH pa;
      load_afrag(pa, Ps[wave], 32, mt * 16, lane);
      O[mt] = WMMA16(pa.v, vb.v, O[mt]);
    }
  }

#pragma unroll
  for (int mt = 0; mt < 2; ++mt)
#pragma unroll
    for (int j = 0; j < 8; ++j) {
      const int nrow = wave * 32 + mt * 16 + j + rofs;
      const float val = O[mt][j] / srun[mt][j];
      ctx[(((size_t)(b * 256 + nrow)) * 64 + t) * 64 + h * 16 + colLane] =
          (_Float16)val;
    }
}

// ---------------------------------------------------------------------------
// Fused temporal attention, one 64-thread block per (b,h,n). T=64, dh=16.
// ---------------------------------------------------------------------------
__global__ __launch_bounds__(64) void attn_temporal_wmma(
    const _Float16* __restrict__ q, const _Float16* __restrict__ k,
    const _Float16* __restrict__ v, _Float16* __restrict__ ctx) {
  __shared__ _Float16 Qs[64 * 32];
  __shared__ _Float16 KsT[32 * 64];
  __shared__ _Float16 Vs[64 * 16];
  __shared__ _Float16 Ps[2][32 * 32];

  const int bidx = blockIdx.x;
  const int n = bidx & 255, h = (bidx >> 8) & 3, b = bidx >> 10;
  const int tid = threadIdx.x, wave = tid >> 5, lane = tid & 31;

  {
    const int tt = tid;
    const size_t base = (((size_t)(b * 256 + n)) * 64 + tt) * 64 + h * 16;
#pragma unroll
    for (int d2 = 0; d2 < 8; ++d2) {
      ((unsigned*)&Qs[tt * 32])[d2]     = *(const unsigned*)(q + base + 2 * d2);
      ((unsigned*)&Qs[tt * 32])[8 + d2] = 0u;
      ((unsigned*)&Vs[tt * 16])[d2]     = *(const unsigned*)(v + base + 2 * d2);
    }
#pragma unroll
    for (int d = 0; d < 16; ++d) {
      KsT[d * 64 + tt]        = k[base + d];
      KsT[(16 + d) * 64 + tt] = (_Float16)0.f;
    }
  }
  __syncthreads();

  FragH qa[2];
  load_afrag(qa[0], Qs, 32, wave * 32, lane);
  load_afrag(qa[1], Qs, 32, wave * 32 + 16, lane);

  v8f O[2] = {v8f_zero(), v8f_zero()};
  float mrun[2][8], srun[2][8];
#pragma unroll
  for (int mt = 0; mt < 2; ++mt)
#pragma unroll
    for (int j = 0; j < 8; ++j) { mrun[mt][j] = -1e30f; srun[mt][j] = 0.f; }

  const float scale = 0.25f;
  const int rofs = (lane < 16) ? 0 : 8;
  const int colLane = lane & 15;

  for (int c = 0; c < 2; ++c) {
    FragH kb0, kb1, vb;
    load_bfrag(kb0, KsT, 64, c * 32, lane);
    load_bfrag(kb1, KsT, 64, c * 32 + 16, lane);
    load_bfrag(vb, Vs + (size_t)c * 32 * 16, 16, 0, lane);

#pragma unroll
    for (int mt = 0; mt < 2; ++mt) {
      v8f S0 = WMMA16(qa[mt].v, kb0.v, v8f_zero());
      v8f S1 = WMMA16(qa[mt].v, kb1.v, v8f_zero());
#pragma unroll
      for (int j = 0; j < 8; ++j) {
        S0[j] *= scale;
        S1[j] *= scale;
        float mx = fmaxf(S0[j], S1[j]);
        for (int off = 8; off; off >>= 1) mx = fmaxf(mx, __shfl_xor(mx, off, 16));
        const float mnew = fmaxf(mrun[mt][j], mx);
        const float f = expf(mrun[mt][j] - mnew);
        mrun[mt][j] = mnew;
        S0[j] = expf(S0[j] - mnew);
        S1[j] = expf(S1[j] - mnew);
        float rs = S0[j] + S1[j];
        for (int off = 8; off; off >>= 1) rs += __shfl_xor(rs, off, 16);
        srun[mt][j] = srun[mt][j] * f + rs;
        O[mt][j] *= f;
        const int rl = mt * 16 + j + rofs;
        Ps[wave][rl * 32 + colLane]      = (_Float16)S0[j];
        Ps[wave][rl * 32 + 16 + colLane] = (_Float16)S1[j];
      }
    }
#pragma unroll
    for (int mt = 0; mt < 2; ++mt) {
      FragH pa;
      load_afrag(pa, Ps[wave], 32, mt * 16, lane);
      O[mt] = WMMA16(pa.v, vb.v, O[mt]);
    }
  }

#pragma unroll
  for (int mt = 0; mt < 2; ++mt)
#pragma unroll
    for (int j = 0; j < 8; ++j) {
      const int trow = wave * 32 + mt * 16 + j + rofs;
      const float val = O[mt][j] / srun[mt][j];
      ctx[(((size_t)(b * 256 + n)) * 64 + trow) * 64 + h * 16 + colLane] =
          (_Float16)val;
    }
}

// ---------------------------------------------------------------------------
// Small kernels
// ---------------------------------------------------------------------------

__global__ void wcvt(const float* __restrict__ s, _Float16* __restrict__ d, int n) {
  int i = blockIdx.x * 256 + threadIdx.x;
  if (i < n) d[i] = (_Float16)s[i];
}

// bias[h][n][m] = rel_pos_emb[adj[n][m]][h]
__global__ void bias_build(const int* __restrict__ adj,
                           const float* __restrict__ rel, float* __restrict__ o) {
  int i = blockIdx.x * 256 + threadIdx.x;  // H*N*N = 262144
  int h = i >> 16, n = (i >> 8) & 255, m = i & 255;
  o[i] = rel[adj[n * 256 + m] * 4 + h];
}

// gap[b,n] = mean |q[b,n,:,:]| over T*C = 4096
__global__ __launch_bounds__(256) void gap_kernel(const float* __restrict__ q,
                                                  float* __restrict__ gap) {
  __shared__ float part[8];
  const float* p = q + (size_t)blockIdx.x * 4096;
  float s = 0.f;
#pragma unroll
  for (int i = 0; i < 16; ++i) s += fabsf(p[threadIdx.x + i * 256]);
  for (int off = 16; off; off >>= 1) s += __shfl_xor(s, off, 32);
  if ((threadIdx.x & 31) == 0) part[threadIdx.x >> 5] = s;
  __syncthreads();
  if (threadIdx.x == 0) {
    float tot = 0.f;
#pragma unroll
    for (int w = 0; w < 8; ++w) tot += part[w];
    gap[blockIdx.x] = tot * (1.f / 4096.f);
  }
}

// gate MLP: thr = gap * sigmoid(relu(bn(gap@fc1+b1))@fc2+b2); one block/batch
__global__ __launch_bounds__(256) void thr_kernel(
    const float* __restrict__ gap, const float* __restrict__ fc1w,
    const float* __restrict__ fc1b, const float* __restrict__ bng,
    const float* __restrict__ bnb, const float* __restrict__ fc2w,
    const float* __restrict__ fc2b, float* __restrict__ thr) {
  __shared__ float g[256], r[256];
  const int b = blockIdx.x, n = threadIdx.x;
  g[n] = gap[b * 256 + n];
  __syncthreads();
  float hsum = fc1b[n];
  for (int kk = 0; kk < 256; ++kk) hsum += g[kk] * fc1w[kk * 256 + n];
  hsum = hsum * (1.f / sqrtf(1.f + 1e-5f)) * bng[n] + bnb[n];
  r[n] = fmaxf(hsum, 0.f);
  __syncthreads();
  float a = fc2b[n];
  for (int kk = 0; kk < 256; ++kk) a += r[kk] * fc2w[kk * 256 + n];
  a = 1.f / (1.f + expf(-a));
  thr[b * 256 + n] = g[n] * a;
}

// x = soft-threshold(query); xt = x + (id_emb + in_deg_emb[g_id] + out_deg_emb[g_od])
__global__ void soft_emb(const float* __restrict__ q, const float* __restrict__ thr,
                         const float* __restrict__ ide, const float* __restrict__ ind,
                         const float* __restrict__ oud, const int* __restrict__ gid,
                         const int* __restrict__ god, float* __restrict__ x,
                         float* __restrict__ xt, _Float16* __restrict__ xth) {
  size_t i = (size_t)blockIdx.x * 256 + threadIdx.x;
  int c = i & 63, n = (i >> 12) & 255, b = i >> 20;
  float qv = q[i];
  float th = thr[b * 256 + n];
  float xv = copysignf(fmaxf(fabsf(qv) - th, 0.f), qv);
  if (qv == 0.f) xv = 0.f;
  float e = ide[n * 64 + c] + ind[gid[n] * 64 + c] + oud[god[n] * 64 + c];
  x[i] = xv;
  float xtv = xv + e;
  xt[i] = xtv;
  xth[i] = (_Float16)xtv;
}

// xt2 = x1 + sinusoid(t,c) + (w_t @ wt_w + wt_b)[b,c]   (f16 output only)
__global__ void temporal_embed(const float* __restrict__ x1,
                               const float* __restrict__ w_t,
                               const float* __restrict__ wtw,
                               const float* __restrict__ wtb,
                               _Float16* __restrict__ o) {
  size_t i = (size_t)blockIdx.x * 256 + threadIdx.x;
  int c = i & 63, t = (i >> 6) & 63, b = i >> 20;
  float angle = (float)t / powf(10000.f, (float)(2 * (c >> 1)) / 64.f);
  float dt = (c & 1) ? cosf(angle) : sinf(angle);
  float dwt = w_t[b * 2 + 0] * wtw[c] + w_t[b * 2 + 1] * wtw[64 + c] + wtb[c];
  o[i] = (_Float16)(x1[i] + dt + dwt);
}

// LayerNorm(a + b) * g + be ; one wave per 64-wide row, 8 rows/block.
__global__ __launch_bounds__(256) void ln_res(
    const float* __restrict__ a, const float* __restrict__ b,
    const float* __restrict__ g, const float* __restrict__ be,
    float* __restrict__ outF, _Float16* __restrict__ outH, int rows) {
  const int wave = threadIdx.x >> 5, lane = threadIdx.x & 31;
  const int row = blockIdx.x * 8 + wave;
  if (row >= rows) return;
  const size_t o = (size_t)row * 64;
  float x0 = a[o + lane] + b[o + lane];
  float x1 = a[o + lane + 32] + b[o + lane + 32];
  float s = x0 + x1;
  for (int off = 16; off; off >>= 1) s += __shfl_xor(s, off, 32);
  const float mu = s * (1.f / 64.f);
  const float d0 = x0 - mu, d1 = x1 - mu;
  float vs = d0 * d0 + d1 * d1;
  for (int off = 16; off; off >>= 1) vs += __shfl_xor(vs, off, 32);
  const float inv = rsqrtf(vs * (1.f / 64.f) + 1e-5f);
  const float y0 = d0 * inv * g[lane] + be[lane];
  const float y1 = d1 * inv * g[lane + 32] + be[lane + 32];
  if (outF) { outF[o + lane] = y0; outF[o + lane + 32] = y1; }
  if (outH) { outH[o + lane] = (_Float16)y0; outH[o + lane + 32] = (_Float16)y1; }
}

// ---------------------------------------------------------------------------
// Host orchestration
// ---------------------------------------------------------------------------
extern "C" void kernel_launch(void* const* d_in, const int* in_sizes, int n_in,
                              void* d_out, int out_size, void* d_ws, size_t ws_size,
                              hipStream_t stream) {
  (void)in_sizes; (void)n_in; (void)out_size; (void)ws_size;
  // Input order: value, key_in, query, t, w_t, adj, g_id, g_od, then params
  // in dict-insertion order.
  const float* query = (const float*)d_in[2];
  const float* w_t   = (const float*)d_in[4];
  const int* adj = (const int*)d_in[5];
  const int* gid = (const int*)d_in[6];
  const int* god = (const int*)d_in[7];
  const float* id_emb = (const float*)d_in[8];
  const float* in_deg = (const float*)d_in[9];
  const float* out_deg = (const float*)d_in[10];
  const float* rel_pos = (const float*)d_in[11];
  const float* s_wq = (const float*)d_in[12];
  const float* s_wk = (const float*)d_in[13];
  const float* s_wv = (const float*)d_in[14];
  const float* s_fow = (const float*)d_in[15];
  const float* s_fob = (const float*)d_in[16];
  const float* s_ff1w = (const float*)d_in[17];
  const float* s_ff1b = (const float*)d_in[18];
  const float* s_ff2w = (const float*)d_in[19];
  const float* s_ff2b = (const float*)d_in[20];
  const float* s_ln1g = (const float*)d_in[21];
  const float* s_ln1b = (const float*)d_in[22];
  const float* s_ln2g = (const float*)d_in[23];
  const float* s_ln2b = (const float*)d_in[24];
  const float* t_wq = (const float*)d_in[25];
  const float* t_wk = (const float*)d_in[26];
  const float* t_wv = (const float*)d_in[27];
  const float* t_fow = (const float*)d_in[28];
  const float* t_fob = (const float*)d_in[29];
  const float* t_ff1w = (const float*)d_in[30];
  const float* t_ff1b = (const float*)d_in[31];
  const float* t_ff2w = (const float*)d_in[32];
  const float* t_ff2b = (const float*)d_in[33];
  const float* t_ln1g = (const float*)d_in[34];
  const float* t_ln1b = (const float*)d_in[35];
  const float* t_ln2g = (const float*)d_in[36];
  const float* t_ln2b = (const float*)d_in[37];
  const float* wt_w = (const float*)d_in[38];
  const float* wt_b = (const float*)d_in[39];
  const float* fc1w = (const float*)d_in[40];
  const float* fc1b = (const float*)d_in[41];
  const float* bng = (const float*)d_in[42];
  const float* bnb = (const float*)d_in[43];
  const float* fc2w = (const float*)d_in[44];
  const float* fc2b = (const float*)d_in[45];
  const float* bl1g = (const float*)d_in[46];
  const float* bl1b = (const float*)d_in[47];
  const float* bl2g = (const float*)d_in[48];
  const float* bl2b = (const float*)d_in[49];

  const int M = 131072;                  // B*N*T
  const size_t ELT = 8388608ull;         // M*C
  const size_t FFE = 33554432ull;        // M*256

  char* w = (char*)d_ws;
  size_t off = 0;
  float* A0 = (float*)(w + off); off += ELT * 4;   // x (soft-thresholded)
  float* A1 = (float*)(w + off); off += ELT * 4;   // xt -> xs -> xT
  float* A2 = (float*)(w + off); off += ELT * 4;   // scratch f32
  float* A3 = (float*)(w + off); off += ELT * 4;   // x1
  _Float16* H0 = (_Float16*)(w + off); off += ELT * 2;  // f16 activation
  _Float16* H1 = (_Float16*)(w + off); off += ELT * 2;  // q
  _Float16* H2 = (_Float16*)(w + off); off += ELT * 2;  // k
  _Float16* H3 = (_Float16*)(w + off); off += ELT * 2;  // v
  _Float16* H4 = (_Float16*)(w + off); off += ELT * 2;  // ctx
  _Float16* H5 = (_Float16*)(w + off); off += FFE * 2;  // ff1
  float* biasH = (float*)(w + off); off += 4ull * 256 * 256 * 4;
  _Float16* WH = (_Float16*)(w + off); off += 98304ull * 2;
  float* gap = (float*)(w + off); off += 2048 * 4;
  float* thr = (float*)(w + off); off += 2048 * 4;

  // f16 weight slots (halves)
  _Float16* Wsq = WH + 0;      _Float16* Wsk = WH + 4096;
  _Float16* Wsv = WH + 8192;   _Float16* Wso = WH + 12288;
  _Float16* Ws1 = WH + 16384;  _Float16* Ws2 = WH + 32768;
  _Float16* Wtq = WH + 49152;  _Float16* Wtk = WH + 53248;
  _Float16* Wtv = WH + 57344;  _Float16* Wto = WH + 61440;
  _Float16* Wt1 = WH + 65536;  _Float16* Wt2 = WH + 81920;

  // --- weights to f16 -------------------------------------------------------
  wcvt<<<16, 256, 0, stream>>>(s_wq, Wsq, 4096);
  wcvt<<<16, 256, 0, stream>>>(s_wk, Wsk, 4096);
  wcvt<<<16, 256, 0, stream>>>(s_wv, Wsv, 4096);
  wcvt<<<16, 256, 0, stream>>>(s_fow, Wso, 4096);
  wcvt<<<64, 256, 0, stream>>>(s_ff1w, Ws1, 16384);
  wcvt<<<64, 256, 0, stream>>>(s_ff2w, Ws2, 16384);
  wcvt<<<16, 256, 0, stream>>>(t_wq, Wtq, 4096);
  wcvt<<<16, 256, 0, stream>>>(t_wk, Wtk, 4096);
  wcvt<<<16, 256, 0, stream>>>(t_wv, Wtv, 4096);
  wcvt<<<16, 256, 0, stream>>>(t_fow, Wto, 4096);
  wcvt<<<64, 256, 0, stream>>>(t_ff1w, Wt1, 16384);
  wcvt<<<64, 256, 0, stream>>>(t_ff2w, Wt2, 16384);
  bias_build<<<1024, 256, 0, stream>>>(adj, rel_pos, biasH);

  // --- soft-threshold gate --------------------------------------------------
  gap_kernel<<<2048, 256, 0, stream>>>(query, gap);
  thr_kernel<<<8, 256, 0, stream>>>(gap, fc1w, fc1b, bng, bnb, fc2w, fc2b, thr);
  soft_emb<<<32768, 256, 0, stream>>>(query, thr, id_emb, in_deg, out_deg, gid,
                                      god, A0, A1, H0);

  dim3 g64(M / 128, 1), g256(M / 128, 4);

  // --- spaformer ------------------------------------------------------------
  gemm_f16_wmma<<<g64, 256, 0, stream>>>(H0, Wsq, nullptr, nullptr, H1, M, 64, 64, 0);
  gemm_f16_wmma<<<g64, 256, 0, stream>>>(H0, Wsk, nullptr, nullptr, H2, M, 64, 64, 0);
  gemm_f16_wmma<<<g64, 256, 0, stream>>>(H0, Wsv, nullptr, nullptr, H3, M, 64, 64, 0);
  attn_spatial_wmma<<<2048, 256, 0, stream>>>(H1, H2, H3, biasH, H4);
  gemm_f16_wmma<<<g64, 256, 0, stream>>>(H4, Wso, s_fob, A2, nullptr, M, 64, 64, 0);
  ln_res<<<M / 8, 256, 0, stream>>>(A2, A1, s_ln1g, s_ln1b, A1, H0, M);   // xs
  gemm_f16_wmma<<<g256, 256, 0, stream>>>(H0, Ws1, s_ff1b, nullptr, H5, M, 64, 256, 1);
  gemm_f16_wmma<<<g64, 256, 0, stream>>>(H5, Ws2, s_ff2b, A2, nullptr, M, 256, 64, 0);
  ln_res<<<M / 8, 256, 0, stream>>>(A2, A1, s_ln2g, s_ln2b, A2, nullptr, M);  // spa
  ln_res<<<M / 8, 256, 0, stream>>>(A2, A0, bl1g, bl1b, A3, nullptr, M);      // x1

  // --- tempformer -----------------------------------------------------------
  temporal_embed<<<32768, 256, 0, stream>>>(A3, w_t, wt_w, wt_b, H0);
  gemm_f16_wmma<<<g64, 256, 0, stream>>>(H0, Wtq, nullptr, nullptr, H1, M, 64, 64, 0);
  gemm_f16_wmma<<<g64, 256, 0, stream>>>(H0, Wtk, nullptr, nullptr, H2, M, 64, 64, 0);
  gemm_f16_wmma<<<g64, 256, 0, stream>>>(H0, Wtv, nullptr, nullptr, H3, M, 64, 64, 0);
  attn_temporal_wmma<<<8192, 64, 0, stream>>>(H1, H2, H3, H4);
  gemm_f16_wmma<<<g64, 256, 0, stream>>>(H4, Wto, t_fob, A2, nullptr, M, 64, 64, 0);
  ln_res<<<M / 8, 256, 0, stream>>>(A2, query, t_ln1g, t_ln1b, A1, H0, M);  // xT
  gemm_f16_wmma<<<g256, 256, 0, stream>>>(H0, Wt1, t_ff1b, nullptr, H5, M, 64, 256, 1);
  gemm_f16_wmma<<<g64, 256, 0, stream>>>(H5, Wt2, t_ff2b, A2, nullptr, M, 256, 64, 0);
  ln_res<<<M / 8, 256, 0, stream>>>(A2, A1, t_ln2g, t_ln2b, A2, nullptr, M);  // temp
  ln_res<<<M / 8, 256, 0, stream>>>(A2, A3, bl2g, bl2b, (float*)d_out, nullptr, M);  // x2
}
